// CommModel_13546326851961
// MI455X (gfx1250) — matrix operand order repaired and checked by reading, not compiled, not run on hardware
//
#include <hip/hip_runtime.h>
#include <hip/hip_bf16.h>
#include <math.h>

// ---------------------------------------------------------------------------
// Problem sizes (fixed by the reference)
// ---------------------------------------------------------------------------
#define BB   64          // batch
#define NN   32          // sequence length
#define D_IN 128         // input dim
#define HH   256         // hidden
#define G4   1024        // 4*H (gates)
#define DOUT 128

typedef __attribute__((ext_vector_type(16))) __bf16 v16bf;
typedef __attribute__((ext_vector_type(8)))  __bf16 v8bf;
typedef __attribute__((ext_vector_type(8)))  float  v8f;

// ---------------------------------------------------------------------------
// A-matrix (16x32 bf16, MxK) fragment load.
// Lanes 0-15: M = lane, K runs {k0..k0+7, k0+16..k0+23} with k0 = kk*32.
// Lanes 16-31: same M = lane-16, runs shifted by +8.
// Each run is 8 bf16 = 16 bytes -> one b128 load.
// ---------------------------------------------------------------------------
template <typename PTR>
__device__ __forceinline__ v16bf load_a_frag(PTR base, int row_stride,
                                             int m, bool hi, int kk) {
    const int k0 = kk * 32 + (hi ? 8 : 0);
    union { v16bf v; v8bf h2[2]; } u;
    u.h2[0] = *(const v8bf*)(base + (size_t)m * row_stride + k0);
    u.h2[1] = *(const v8bf*)(base + (size_t)m * row_stride + k0 + 16);
    return u.v;
}

// ---------------------------------------------------------------------------
// B-matrix (32x16 bf16, KxN) fragment load from a row-major weight W[n][k]
// (B[k][n] = W[n][k]).  Lane holds column n = (lane&15); lanes 0-15 carry
// K = kk*32 + {0..7, 16..23}, lanes 16-31 carry K = kk*32 + {8..15, 24..31}:
// two 8-half (16B, aligned) loads per lane, matching the A fragment.
// ---------------------------------------------------------------------------
__device__ __forceinline__ v16bf load_b_frag(const __bf16* __restrict__ w,
                                             int row_stride, int n, bool hi,
                                             int kk) {
    const int k0 = kk * 32 + (hi ? 8 : 0);
    union { v16bf v; v8bf h2[2]; } u;
    u.h2[0] = *(const v8bf*)(w + (size_t)n * row_stride + k0);
    u.h2[1] = *(const v8bf*)(w + (size_t)n * row_stride + k0 + 16);
    return u.v;
}

__device__ __forceinline__ v8f wmma_bf16(v16bf a, v16bf b, v8f c) {
    return __builtin_amdgcn_wmma_f32_16x16x32_bf16(
        /*neg_a=*/false, a, /*neg_b=*/false, b,
        /*c_mod=*/(short)0, c, /*reuse_a=*/false, /*reuse_b=*/false);
}

__device__ __forceinline__ float sigmoidf_fast(float x) {
    return 1.0f / (1.0f + __expf(-x));
}

// ---------------------------------------------------------------------------
// Kernel 1: f32 -> bf16 conversion (weights and x)
// ---------------------------------------------------------------------------
__global__ void f32_to_bf16_kernel(const float* __restrict__ in,
                                   __bf16* __restrict__ out, int n) {
    int i = blockIdx.x * blockDim.x + threadIdx.x;
    if (i < n) out[i] = (__bf16)in[i];
}

// ---------------------------------------------------------------------------
// Kernel 2: Xproj[s2, n] = sum_k x_bf[s2, k] * W_ih[n, k]   (s2 = b*NN + t)
// A = x (2048 x 128), B = W_ih^T (128 x 1024).  One block per 16-row tile,
// 8 waves, each wave covers 8 n-tiles, K = 4 fragments.
// ---------------------------------------------------------------------------
__global__ __launch_bounds__(256)
void xproj_kernel(const __bf16* __restrict__ xbf,    // (BB*NN, D_IN)
                  const __bf16* __restrict__ wih,    // (G4, D_IN) row-major
                  float* __restrict__ xproj) {       // (BB*NN, G4)
    const int mtile = blockIdx.x;                    // 0..127
    const int tid  = threadIdx.x;
    const int lane = tid & 31;
    const int wave = tid >> 5;                       // 0..7
    const bool hi  = lane >= 16;
    const int  ln  = lane & 15;

    v16bf afrag[4];
#pragma unroll
    for (int kk = 0; kk < 4; ++kk)
        afrag[kk] = load_a_frag(xbf, D_IN, mtile * 16 + ln, hi, kk);

    for (int nn = 0; nn < 8; ++nn) {                 // this wave's n-tiles
        const int nt = wave * 8 + nn;
        const int n  = nt * 16 + ln;
        v8f acc = {};
#pragma unroll
        for (int kk = 0; kk < 4; ++kk)
            acc = wmma_bf16(afrag[kk], load_b_frag(wih, D_IN, n, hi, kk), acc);
#pragma unroll
        for (int v = 0; v < 8; ++v) {
            const int m = v + (hi ? 8 : 0);
            xproj[(size_t)(mtile * 16 + m) * G4 + n] = acc[v];
        }
    }
}

// ---------------------------------------------------------------------------
// Kernel 3: diagonal bidirectional LSTM scan.
// Block = (dir, grp, q): 16 sequences b = grp*16+m sharing row q.
//   forward:  t = 0..q      (q+1 steps)   -> h at t=q
//   reverse:  t = 31..q     (32-q steps)  -> h at t=q
// Per step: g = h @ W_hh^T via WMMA (h in LDS as bf16), then per-lane cell
// update with mask * Xproj + bias, c kept in registers.
// Wave w owns gate tiles {gate*16 + 2w, gate*16 + 2w + 1}, so i/f/g/o for a
// given (seq m, hidden j) all live in the same lane's accumulators.
// ---------------------------------------------------------------------------
__global__ __launch_bounds__(256)
void lstm_diag_kernel(const float* __restrict__ mask,     // (BB, NN, NN)
                      const float* __restrict__ bias_f,
                      const float* __restrict__ bias_r,
                      const float* __restrict__ xproj_f,  // (BB*NN, G4)
                      const float* __restrict__ xproj_r,
                      const __bf16* __restrict__ whh_f,   // (G4, HH) bf16
                      const __bf16* __restrict__ whh_r,
                      __bf16* __restrict__ y) {           // (BB*NN, 2*HH)
    __shared__ __bf16 h_lds[16][HH];                      // 8 KB

    const int blk = blockIdx.x;                           // 0..255
    const int dir = blk >> 7;                             // 0 fwd, 1 rev
    const int grp = (blk >> 5) & 3;                       // 0..3 (group of 16 b)
    const int q   = blk & 31;

    const int tid  = threadIdx.x;
    const int lane = tid & 31;
    const int wave = tid >> 5;
    const bool hi  = lane >= 16;
    const int  ln  = lane & 15;

    const float*  xproj = dir ? xproj_r : xproj_f;
    const __bf16* whh   = dir ? whh_r   : whh_f;
    const float*  bias  = dir ? bias_r  : bias_f;

    // zero-init h state
    for (int i = tid; i < 16 * HH; i += 256)
        (&h_lds[0][0])[i] = (__bf16)0.0f;
    float c[2][8];
#pragma unroll
    for (int s = 0; s < 2; ++s)
#pragma unroll
        for (int v = 0; v < 8; ++v) c[s][v] = 0.0f;
    __syncthreads();

    const int jt0    = wave * 2;                          // first j-tile of wave
    const int nsteps = dir ? (NN - q) : (q + 1);
    int t            = dir ? (NN - 1) : 0;
    const int tstep  = dir ? -1 : 1;

    for (int it = 0; it < nsteps; ++it, t += tstep) {
        // ---- recurrent GEMM: g = h @ W_hh^T --------------------------------
        v16bf afrag[8];
#pragma unroll
        for (int kk = 0; kk < 8; ++kk)
            afrag[kk] = load_a_frag(&h_lds[0][0], HH, ln, hi, kk);
        __syncthreads();   // all reads of h_lds done before rewrites below

        v8f acc[4][2];
#pragma unroll
        for (int gate = 0; gate < 4; ++gate) {
#pragma unroll
            for (int s = 0; s < 2; ++s) {
                const int n = gate * HH + (jt0 + s) * 16 + ln;
                v8f a = {};
#pragma unroll
                for (int kk = 0; kk < 8; ++kk)
                    a = wmma_bf16(afrag[kk], load_b_frag(whh, HH, n, hi, kk), a);
                acc[gate][s] = a;
            }
        }

        // ---- cell update ---------------------------------------------------
#pragma unroll
        for (int s = 0; s < 2; ++s) {
            const int j = (jt0 + s) * 16 + ln;
#pragma unroll
            for (int v = 0; v < 8; ++v) {
                const int m = v + (hi ? 8 : 0);
                const int b = grp * 16 + m;
                const float mk = mask[((size_t)b * NN + q) * NN + t];
                const float* xp = xproj + ((size_t)b * NN + t) * G4;
                const float gi = acc[0][s][v] + bias[0 * HH + j] + mk * xp[0 * HH + j];
                const float gf = acc[1][s][v] + bias[1 * HH + j] + mk * xp[1 * HH + j];
                const float gg = acc[2][s][v] + bias[2 * HH + j] + mk * xp[2 * HH + j];
                const float go = acc[3][s][v] + bias[3 * HH + j] + mk * xp[3 * HH + j];
                const float cc = sigmoidf_fast(gf) * c[s][v] +
                                 sigmoidf_fast(gi) * tanhf(gg);
                c[s][v] = cc;
                const float hh = sigmoidf_fast(go) * tanhf(cc);
                h_lds[m][j] = (__bf16)hh;
            }
        }
        __syncthreads();   // h ready for next step's A-fragment loads
    }

    // final h (t == q) -> y[b*NN+q, dir*HH + j]
#pragma unroll
    for (int s = 0; s < 2; ++s) {
        const int j = (jt0 + s) * 16 + ln;
#pragma unroll
        for (int v = 0; v < 8; ++v) {
            const int m = v + (hi ? 8 : 0);
            const int b = grp * 16 + m;
            y[(size_t)(b * NN + q) * (2 * HH) + dir * HH + j] = h_lds[m][j];
        }
    }
}

// ---------------------------------------------------------------------------
// Kernel 4: out = relu(y @ W_fc^T + b_fc)   (2048x512)@(512x128) via WMMA
// ---------------------------------------------------------------------------
__global__ __launch_bounds__(256)
void fc_kernel(const __bf16* __restrict__ ybf,   // (BB*NN, 2*HH)
               const __bf16* __restrict__ wfc,   // (DOUT, 2*HH) row-major
               const float* __restrict__ b_fc,   // (DOUT)
               float* __restrict__ out) {        // (BB*NN, DOUT)
    const int mtile = blockIdx.x;                // 0..127
    const int tid  = threadIdx.x;
    const int lane = tid & 31;
    const int wave = tid >> 5;                   // = n-tile (0..7)
    const bool hi  = lane >= 16;
    const int  ln  = lane & 15;

    const int n = wave * 16 + ln;
    v8f acc = {};
    for (int kk = 0; kk < 16; ++kk) {
        v16bf a = load_a_frag(ybf, 2 * HH, mtile * 16 + ln, hi, kk);
        acc = wmma_bf16(a, load_b_frag(wfc, 2 * HH, n, hi, kk), acc);
    }
    const float bo = b_fc[n];
#pragma unroll
    for (int v = 0; v < 8; ++v) {
        const int m = v + (hi ? 8 : 0);
        out[(size_t)(mtile * 16 + m) * DOUT + n] = fmaxf(acc[v] + bo, 0.0f);
    }
}

// ---------------------------------------------------------------------------
// Host launcher
// ---------------------------------------------------------------------------
extern "C" void kernel_launch(void* const* d_in, const int* in_sizes, int n_in,
                              void* d_out, int out_size, void* d_ws, size_t ws_size,
                              hipStream_t stream) {
    const float* x      = (const float*)d_in[0];   // (64,32,128)
    const float* mask   = (const float*)d_in[1];   // (64,32,32)
    const float* W_ih_f = (const float*)d_in[2];   // (1024,128)
    const float* W_hh_f = (const float*)d_in[3];   // (1024,256)
    const float* b_f    = (const float*)d_in[4];   // (1024)
    const float* W_ih_r = (const float*)d_in[5];
    const float* W_hh_r = (const float*)d_in[6];
    const float* b_r    = (const float*)d_in[7];
    const float* W_fc   = (const float*)d_in[8];   // (128,512)
    const float* b_fc   = (const float*)d_in[9];   // (128)
    float* out          = (float*)d_out;           // (64,32,128)

    // ---- workspace carve-out (byte offsets, all 32B-aligned) --------------
    const size_t N_X   = (size_t)BB * NN * D_IN;   // 262144
    const size_t N_WIH = (size_t)G4 * D_IN;        // 131072
    const size_t N_WHH = (size_t)G4 * HH;          // 262144
    const size_t N_WFC = (size_t)DOUT * 2 * HH;    // 65536
    const size_t N_Y   = (size_t)BB * NN * 2 * HH; // 1048576
    const size_t N_XP  = (size_t)BB * NN * G4;     // 2097152

    char* ws = (char*)d_ws;
    size_t off = 0;
    auto carve = [&](size_t bytes) { char* p = ws + off; off += bytes; return p; };
    __bf16* x_bf    = (__bf16*)carve(N_X   * 2);
    __bf16* wihf_bf = (__bf16*)carve(N_WIH * 2);
    __bf16* wihr_bf = (__bf16*)carve(N_WIH * 2);
    __bf16* whhf_bf = (__bf16*)carve(N_WHH * 2);
    __bf16* whhr_bf = (__bf16*)carve(N_WHH * 2);
    __bf16* wfc_bf  = (__bf16*)carve(N_WFC * 2);
    __bf16* y_bf    = (__bf16*)carve(N_Y   * 2);
    float*  xproj_f = (float*)carve(N_XP * 4);
    float*  xproj_r = (float*)carve(N_XP * 4);
    if (off > ws_size) return;   // workspace too small: nothing sane to do

    const int CT = 256;
    auto cvt = [&](const float* src, __bf16* dst, size_t n) {
        f32_to_bf16_kernel<<<(unsigned)((n + CT - 1) / CT), CT, 0, stream>>>(src, dst, (int)n);
    };
    cvt(x,      x_bf,    N_X);
    cvt(W_ih_f, wihf_bf, N_WIH);
    cvt(W_ih_r, wihr_bf, N_WIH);
    cvt(W_hh_f, whhf_bf, N_WHH);
    cvt(W_hh_r, whhr_bf, N_WHH);
    cvt(W_fc,   wfc_bf,  N_WFC);

    xproj_kernel<<<128, 256, 0, stream>>>(x_bf, wihf_bf, xproj_f);
    xproj_kernel<<<128, 256, 0, stream>>>(x_bf, wihr_bf, xproj_r);

    lstm_diag_kernel<<<256, 256, 0, stream>>>(mask, b_f, b_r, xproj_f, xproj_r,
                                              whhf_bf, whhr_bf, y_bf);

    fc_kernel<<<128, 256, 0, stream>>>(y_bf, wfc_bf, b_fc, out);
}
